// MemConv_91233695301986
// MI455X (gfx1250) — matrix operand order repaired and checked by reading, not compiled, not run on hardware
//
#include <hip/hip_runtime.h>

#define DEV_INLINE __device__ __forceinline__

constexpr int Bn = 16, Sn = 512, Fn = 128, Hn = 4, Kn = 1024, DOUTn = 128;
constexpr float EPSF = 1e-8f;

typedef __attribute__((ext_vector_type(16))) __bf16 v16bf;
typedef __attribute__((ext_vector_type(8)))  __bf16 v8bf;
typedef __attribute__((ext_vector_type(8)))  float  v8f;

DEV_INLINE v8f wmma_bf16(v16bf a, v16bf b, v8f c) {
  // D = A(16x32 bf16) x B(32x16 bf16) + C(16x16 f32)
  return __builtin_amdgcn_wmma_f32_16x16x32_bf16(false, a, false, b, (short)0, c,
                                                 false, false);
}

// split f32 -> bf16 hi + bf16 lo correction (bf16x3 emulation of f32 GEMM)
DEV_INLINE void split2(float x, __bf16 &hi, __bf16 &lo) {
  __bf16 h = (__bf16)x;
  hi = h;
  lo = (__bf16)(x - (float)h);
}

DEV_INLINE v16bf cat8(v8bf a, v8bf b) {
  return __builtin_shufflevector(a, b, 0, 1, 2, 3, 4, 5, 6, 7, 8, 9, 10, 11, 12, 13, 14, 15);
}

// A-fragment inner-dim index for 16-bit A 16x32: k = hlf*16 + j  (16 contiguous,
// matching B).
// B-fragment inner-dim index for 16-bit B 32x16: k = hlf*16 + j  (16 contiguous).

// ---------------------------------------------------------------------------
// Kernel 0: row sums of squares: q2[B*S], k2[H*K]
// ---------------------------------------------------------------------------
__global__ void k_sumsq(const float* __restrict__ q, const float* __restrict__ keys,
                        float* __restrict__ q2, float* __restrict__ k2) {
  int idx = blockIdx.x * blockDim.x + threadIdx.x;
  if (idx < Bn * Sn) {
    const float* r = q + (size_t)idx * Fn;
    float s = 0.f;
    for (int f = 0; f < Fn; ++f) s += r[f] * r[f];
    q2[idx] = s;
  } else if (idx < Bn * Sn + Hn * Kn) {
    int j = idx - Bn * Sn;
    const float* r = keys + (size_t)j * Fn;
    float s = 0.f;
    for (int f = 0; f < Fn; ++f) s += r[f] * r[f];
    k2[j] = s;
  }
}

// ---------------------------------------------------------------------------
// Prep: one-shot f32 -> (bf16 hi, bf16 lo) splits. Bandwidth-trivial (~6 MB),
// removes per-element cvt work from every GEMM inner loop.
// ---------------------------------------------------------------------------
__global__ void k_split(const float* __restrict__ x, __bf16* __restrict__ hi,
                        __bf16* __restrict__ lo, int n) {
  int idx = blockIdx.x * blockDim.x + threadIdx.x;
  if (idx >= n) return;
  __bf16 h, l;
  split2(x[idx], h, l);
  hi[idx] = h;
  lo[idx] = l;
}

// q [b][s][f] -> qT hi/lo [b][f][s]  (so k_vout phase-1 B fragments are contiguous)
__global__ void k_splitT(const float* __restrict__ q, __bf16* __restrict__ th,
                         __bf16* __restrict__ tl) {
  int idx = blockIdx.x * blockDim.x + threadIdx.x;   // over B*F*S
  if (idx >= Bn * Fn * Sn) return;
  int s  = idx & (Sn - 1);
  int bf = idx >> 9;
  int f  = bf & (Fn - 1);
  int b  = bf >> 7;
  __bf16 h, l;
  split2(q[((size_t)(b * Sn + s)) * Fn + f], h, l);
  th[idx] = h;
  tl[idx] = l;
}

// ---------------------------------------------------------------------------
// Kernel 1: fused qk GEMM (bf16x3 WMMA) + distance kernel + per-h L1 norm +
//           conv combine over h + softmax over K + mask  ->  c (B,S,K)
// One workgroup per (b, 16-row s-tile). 8 waves; wave w owns column tiles
// [w*8, w*8+8). B fragments are direct v16bf loads from pre-split keys.
// ---------------------------------------------------------------------------
__launch_bounds__(256)
__global__ void k_attn(const float* __restrict__ q, const int* __restrict__ mask,
                       const __bf16* __restrict__ keys_hi, const __bf16* __restrict__ keys_lo,
                       const float* __restrict__ conv_w,
                       const float* __restrict__ q2, const float* __restrict__ k2,
                       float* __restrict__ cbuf) {
  const int b  = blockIdx.x >> 5;   // S/16 = 32 tiles
  const int st = blockIdx.x & 31;
  const int s0 = st * 16;

  __shared__ float qt[16 * Fn];   // 8 KB q tile (f32)
  __shared__ float red[8 * 16];   // per-wave row partials
  __shared__ float rowA[16];
  __shared__ float rowB[16];
  __shared__ float q2s[16];
  __shared__ float mk[16];

  const int tid = threadIdx.x;
  for (int i = tid; i < 16 * Fn; i += 256) {
    int m = i >> 7, f = i & (Fn - 1);
    qt[i] = q[((size_t)(b * Sn + s0 + m)) * Fn + f];
  }
  if (tid < 16) {
    q2s[tid] = q2[b * Sn + s0 + tid];
    mk[tid]  = (float)mask[b * Sn + s0 + tid];
  }
  __syncthreads();

  const int lane = tid & 31, wave = tid >> 5;
  const int hlf = lane >> 4, n16 = lane & 15;

  // Precompute A fragments (q tile) once; reused across 4 heads x 8 tiles.
  v16bf a_hi[4], a_lo[4];
#pragma unroll
  for (int fc = 0; fc < 4; ++fc) {
#pragma unroll
    for (int j = 0; j < 16; ++j) {
      int f = fc * 32 + hlf * 16 + j;
      __bf16 h, l;
      split2(qt[n16 * Fn + f], h, l);
      a_hi[fc][j] = h;
      a_lo[fc][j] = l;
    }
  }

  float acc[8][8];
#pragma unroll
  for (int t = 0; t < 8; ++t)
#pragma unroll
    for (int i = 0; i < 8; ++i) acc[t][i] = 0.f;

  for (int h = 0; h < Hn; ++h) {
    float tmp[8][8];
    float rs[8];
#pragma unroll
    for (int i = 0; i < 8; ++i) rs[i] = 0.f;

#pragma unroll
    for (int t = 0; t < 8; ++t) {
      const int nt  = wave * 8 + t;
      const int col = nt * 16 + n16;
      v8f cf = {0.f, 0.f, 0.f, 0.f, 0.f, 0.f, 0.f, 0.f};
#pragma unroll
      for (int fc = 0; fc < 4; ++fc) {
        const size_t off = ((size_t)h * Kn + col) * Fn + fc * 32 + hlf * 16;
        v16bf b_hi = *(const v16bf*)(keys_hi + off);   // 16 contiguous bf16 (32 B)
        v16bf b_lo = *(const v16bf*)(keys_lo + off);
        if (t < 7) __builtin_prefetch(keys_hi + off + 16 * Fn, 0, 3);  // next tile
        cf = wmma_bf16(a_hi[fc], b_hi, cf);
        cf = wmma_bf16(a_hi[fc], b_lo, cf);
        cf = wmma_bf16(a_lo[fc], b_hi, cf);
      }
#pragma unroll
      for (int i = 0; i < 8; ++i) {
        const int m = i + 8 * hlf;
        float d2 = q2s[m] + k2[h * Kn + col] - 2.0f * cf[i];
        d2 = fmaxf(d2, 0.f);
        // (1 + (sqrt(d2)*m)^2 / TAU)^(-(0.5*TAU+0.5)) with TAU=1 -> 1/(1+d2*m)
        float cv = 1.0f / (1.0f + d2 * mk[m]);
        tmp[t][i] = cv;
        rs[i] += cv;
      }
    }
    // reduce row sums across the 16 lanes of each half
#pragma unroll
    for (int xm = 1; xm < 16; xm <<= 1)
#pragma unroll
      for (int i = 0; i < 8; ++i) rs[i] += __shfl_xor(rs[i], xm, 32);
    if (n16 == 0)
#pragma unroll
      for (int i = 0; i < 8; ++i) red[wave * 16 + hlf * 8 + i] = rs[i];
    __syncthreads();
    if (tid < 16) {
      float s = 0.f;
      for (int w = 0; w < 8; ++w) s += red[w * 16 + tid];
      rowA[tid] = s;
    }
    __syncthreads();
    const float cw = conv_w[h];
#pragma unroll
    for (int t = 0; t < 8; ++t)
#pragma unroll
      for (int i = 0; i < 8; ++i) {
        const int m = i + 8 * hlf;
        acc[t][i] += cw * tmp[t][i] * mk[m] / rowA[m];
      }
    __syncthreads();
  }

  // ---- softmax over K per row, then mask, then store ----
  float rmx[8];
#pragma unroll
  for (int i = 0; i < 8; ++i) rmx[i] = -1e30f;
#pragma unroll
  for (int t = 0; t < 8; ++t)
#pragma unroll
    for (int i = 0; i < 8; ++i) rmx[i] = fmaxf(rmx[i], acc[t][i]);
#pragma unroll
  for (int xm = 1; xm < 16; xm <<= 1)
#pragma unroll
    for (int i = 0; i < 8; ++i) rmx[i] = fmaxf(rmx[i], __shfl_xor(rmx[i], xm, 32));
  if (n16 == 0)
#pragma unroll
    for (int i = 0; i < 8; ++i) red[wave * 16 + hlf * 8 + i] = rmx[i];
  __syncthreads();
  if (tid < 16) {
    float s = -1e30f;
    for (int w = 0; w < 8; ++w) s = fmaxf(s, red[w * 16 + tid]);
    rowA[tid] = s;
  }
  __syncthreads();

  float rse[8];
#pragma unroll
  for (int i = 0; i < 8; ++i) rse[i] = 0.f;
#pragma unroll
  for (int t = 0; t < 8; ++t)
#pragma unroll
    for (int i = 0; i < 8; ++i) {
      const int m = i + 8 * hlf;
      float e = __expf(acc[t][i] - rowA[m]);
      acc[t][i] = e;
      rse[i] += e;
    }
#pragma unroll
  for (int xm = 1; xm < 16; xm <<= 1)
#pragma unroll
    for (int i = 0; i < 8; ++i) rse[i] += __shfl_xor(rse[i], xm, 32);
  if (n16 == 0)
#pragma unroll
    for (int i = 0; i < 8; ++i) red[wave * 16 + hlf * 8 + i] = rse[i];
  __syncthreads();
  if (tid < 16) {
    float s = 0.f;
    for (int w = 0; w < 8; ++w) s += red[w * 16 + tid];
    rowB[tid] = s;
  }
  __syncthreads();

#pragma unroll
  for (int t = 0; t < 8; ++t)
#pragma unroll
    for (int i = 0; i < 8; ++i) {
      const int m   = i + 8 * hlf;
      const int col = (wave * 8 + t) * 16 + n16;
      cbuf[((size_t)(b * Sn + s0 + m)) * Kn + col] = acc[t][i] / rowB[m] * mk[m];
    }
}

// ---------------------------------------------------------------------------
// Kernel 2: cn[b][k] = sum_s c + EPS
// ---------------------------------------------------------------------------
__global__ void k_cn(const float* __restrict__ cbuf, float* __restrict__ cn) {
  int idx = blockIdx.x * blockDim.x + threadIdx.x;
  if (idx >= Bn * Kn) return;
  int b = idx / Kn, k = idx % Kn;
  const float* p = cbuf + (size_t)b * Sn * Kn + k;
  float s = 0.f;
  for (int ss = 0; ss < Sn; ++ss) s += p[(size_t)ss * Kn];
  cn[idx] = s + EPSF;
}

// ---------------------------------------------------------------------------
// Kernel 3: per (b,s) row: pn then KL partial. One block per row.
// ---------------------------------------------------------------------------
__launch_bounds__(256)
__global__ void k_pnkl(const float* __restrict__ cbuf, const float* __restrict__ cn,
                       float* __restrict__ klpart) {
  const int row = blockIdx.x;   // b*S + s
  const int b   = row / Sn;
  const float* cr  = cbuf + (size_t)row * Kn;
  const float* cnb = cn + (size_t)b * Kn;
  __shared__ float sm[256];
  const int tid = threadIdx.x;

  float u[4], cl[4];
  float ps = 0.f;
#pragma unroll
  for (int j = 0; j < 4; ++j) {
    int k = tid + j * 256;
    float cv = cr[k];
    float uu = cv * cv / cnb[k];
    cl[j] = cv;
    u[j]  = uu;
    ps += uu;
  }
  sm[tid] = ps;
  __syncthreads();
  for (int off = 128; off > 0; off >>= 1) {
    if (tid < off) sm[tid] += sm[tid + off];
    __syncthreads();
  }
  const float pn = sm[0] + EPSF;
  __syncthreads();

  float kls = 0.f;
#pragma unroll
  for (int j = 0; j < 4; ++j) {
    float p = u[j] / pn;
    kls += p * (logf(p + EPSF) - logf(cl[j] + EPSF));
  }
  sm[tid] = kls;
  __syncthreads();
  for (int off = 128; off > 0; off >>= 1) {
    if (tid < off) sm[tid] += sm[tid + off];
    __syncthreads();
  }
  if (tid == 0) klpart[row] = sm[0];
}

__global__ void k_klfin(const float* __restrict__ klpart, float* __restrict__ out_kl) {
  __shared__ float sm[256];
  const int tid = threadIdx.x;
  float s = 0.f;
  for (int i = tid; i < Bn * Sn; i += 256) s += klpart[i];
  sm[tid] = s;
  __syncthreads();
  for (int off = 128; off > 0; off >>= 1) {
    if (tid < off) sm[tid] += sm[tid + off];
    __syncthreads();
  }
  if (tid == 0) out_kl[0] = sm[0] * 100.0f;
}

// ---------------------------------------------------------------------------
// Kernel 4: fused  v = c^T q  then  out = v lin_w^T + bias, leaky ReLU.
// One workgroup per (b, 16-row k-tile). Runtime A operands (c tile, v tile)
// are split to bf16 hi/lo ONCE into LDS, read as 16 B ds loads per fragment.
// B operands come pre-split from global (contiguous v16bf loads).
// ---------------------------------------------------------------------------
__launch_bounds__(256)
__global__ void k_vout(const float* __restrict__ cbuf,
                       const __bf16* __restrict__ qT_hi, const __bf16* __restrict__ qT_lo,
                       const __bf16* __restrict__ lw_hi, const __bf16* __restrict__ lw_lo,
                       const float* __restrict__ lin_b, float* __restrict__ out) {
  const int b  = blockIdx.x >> 6;   // K/16 = 64 tiles
  const int kt = blockIdx.x & 63;
  const int k0 = kt * 16;

  __shared__ __align__(32) __bf16 ct_hi[16 * Sn];   // 16 KB
  __shared__ __align__(32) __bf16 ct_lo[16 * Sn];   // 16 KB
  __shared__ float vt[16 * Fn];                     //  8 KB
  __shared__ __align__(32) __bf16 vt_hi[16 * Fn];   //  4 KB
  __shared__ __align__(32) __bf16 vt_lo[16 * Fn];   //  4 KB

  const int tid = threadIdx.x;
  for (int i = tid; i < 16 * Sn; i += 256) {
    int s = i >> 4, kk = i & 15;
    __bf16 h, l;
    split2(cbuf[((size_t)(b * Sn + s)) * Kn + k0 + kk], h, l);
    ct_hi[kk * Sn + s] = h;
    ct_lo[kk * Sn + s] = l;
  }
  __syncthreads();

  const int lane = tid & 31, wave = tid >> 5;
  const int hlf = lane >> 4, n16 = lane & 15;

  // ---- Phase 1: v[16 x F] = ct[16 x S] * q_b[S x F], wave slice f = [wave*16,+16)
  {
    v8f accv = {0.f, 0.f, 0.f, 0.f, 0.f, 0.f, 0.f, 0.f};
    const int fcol = wave * 16 + n16;
    for (int sc = 0; sc < Sn / 32; ++sc) {
      const int sbase = sc * 32;
      const __bf16* ah = ct_hi + n16 * Sn + sbase;
      const __bf16* al = ct_lo + n16 * Sn + sbase;
      v16bf a_hi = *(const v16bf*)(ah + hlf * 16);
      v16bf a_lo = *(const v16bf*)(al + hlf * 16);
      const size_t qo = ((size_t)(b * Fn + fcol)) * Sn + sbase + hlf * 16;
      v16bf b_hi = *(const v16bf*)(qT_hi + qo);   // contiguous 32 B
      v16bf b_lo = *(const v16bf*)(qT_lo + qo);
      accv = wmma_bf16(a_hi, b_hi, accv);
      accv = wmma_bf16(a_hi, b_lo, accv);
      accv = wmma_bf16(a_lo, b_hi, accv);
    }
#pragma unroll
    for (int i = 0; i < 8; ++i) vt[(i + 8 * hlf) * Fn + wave * 16 + n16] = accv[i];
  }
  __syncthreads();

  // split v tile once
  for (int i = tid; i < 16 * Fn; i += 256) {
    __bf16 h, l;
    split2(vt[i], h, l);
    vt_hi[i] = h;
    vt_lo[i] = l;
  }
  __syncthreads();

  // ---- Phase 2: out[16 x DOUT] = vt[16 x F] * lin_w^T[F x DOUT], wave slice o
  {
    v8f acco = {0.f, 0.f, 0.f, 0.f, 0.f, 0.f, 0.f, 0.f};
    const int o = wave * 16 + n16;
#pragma unroll
    for (int fc = 0; fc < 4; ++fc) {
      const __bf16* ah = vt_hi + n16 * Fn + fc * 32;
      const __bf16* al = vt_lo + n16 * Fn + fc * 32;
      v16bf a_hi = *(const v16bf*)(ah + hlf * 16);
      v16bf a_lo = *(const v16bf*)(al + hlf * 16);
      const size_t wo = (size_t)o * Fn + fc * 32 + hlf * 16;
      v16bf b_hi = *(const v16bf*)(lw_hi + wo);
      v16bf b_lo = *(const v16bf*)(lw_lo + wo);
      acco = wmma_bf16(a_hi, b_hi, acco);
      acco = wmma_bf16(a_hi, b_lo, acco);
      acco = wmma_bf16(a_lo, b_hi, acco);
    }
    const float bias = lin_b[o];
#pragma unroll
    for (int i = 0; i < 8; ++i) {
      const int m = i + 8 * hlf;
      float v = acco[i] + bias;
      v = (v >= 0.f) ? v : 0.01f * v;
      out[((size_t)(b * Kn) + k0 + m) * DOUTn + o] = v;
    }
  }
}

// ---------------------------------------------------------------------------
extern "C" void kernel_launch(void* const* d_in, const int* in_sizes, int n_in,
                              void* d_out, int out_size, void* d_ws, size_t ws_size,
                              hipStream_t stream) {
  (void)in_sizes; (void)n_in; (void)out_size; (void)ws_size;
  const float* q      = (const float*)d_in[0];
  const int*   mask   = (const int*)d_in[1];
  const float* keys   = (const float*)d_in[2];
  const float* conv_w = (const float*)d_in[3];
  const float* lin_w  = (const float*)d_in[4];
  const float* lin_b  = (const float*)d_in[5];
  float* out = (float*)d_out;

  float* ws     = (float*)d_ws;
  float* q2     = ws;                       // 8192
  float* k2     = q2 + Bn * Sn;             // 4096
  float* cn     = k2 + Hn * Kn;             // 16384
  float* klpart = cn + Bn * Kn;             // 8192
  float* cbuf   = klpart + Bn * Sn;         // 8388608
  float* bf     = cbuf + (size_t)Bn * Sn * Kn;

  const int NKEY = Hn * Kn * Fn;            // 524288
  const int NQ   = Bn * Fn * Sn;            // 1048576
  const int NW   = DOUTn * Fn;              // 16384
  __bf16* keys_hi = (__bf16*)bf;
  __bf16* keys_lo = keys_hi + NKEY;
  __bf16* qT_hi   = keys_lo + NKEY;
  __bf16* qT_lo   = qT_hi + NQ;
  __bf16* lw_hi   = qT_lo + NQ;
  __bf16* lw_lo   = lw_hi + NW;

  k_sumsq<<<(Bn * Sn + Hn * Kn + 255) / 256, 256, 0, stream>>>(q, keys, q2, k2);
  k_split<<<(NKEY + 255) / 256, 256, 0, stream>>>(keys, keys_hi, keys_lo, NKEY);
  k_split<<<(NW + 255) / 256, 256, 0, stream>>>(lin_w, lw_hi, lw_lo, NW);
  k_splitT<<<(NQ + 255) / 256, 256, 0, stream>>>(q, qT_hi, qT_lo);

  k_attn<<<Bn * (Sn / 16), 256, 0, stream>>>(q, mask, keys_hi, keys_lo, conv_w,
                                             q2, k2, cbuf);
  k_cn<<<(Bn * Kn + 255) / 256, 256, 0, stream>>>(cbuf, cn);
  k_pnkl<<<Bn * Sn, 256, 0, stream>>>(cbuf, cn, klpart);
  k_klfin<<<1, 256, 0, stream>>>(klpart, out + (size_t)Bn * Kn * DOUTn);
  k_vout<<<Bn * (Kn / 16), 256, 0, stream>>>(cbuf, qT_hi, qT_lo, lw_hi, lw_lo,
                                             lin_b, out);
}
